// ElectronegativityNet_38920993636805
// MI455X (gfx1250) — matrix operand order repaired and checked by reading, not compiled, not run on hardware
//
#include <hip/hip_runtime.h>
#include <hip/hip_bf16.h>

// ---------------------------------------------------------------------------
// ElectronegativityNet MoE: expert-routed 2-layer MLP + linear head.
// Bucket atoms by expert (regions padded to 256), then 16-atom tiles through
// v_wmma_f32_16x16x32_f16 with split-precision (hi/lo f16) operands.
// Each block stages one expert's weights in LDS once and processes 4 chunks
// of 64 atoms (4 waves x 16-row tiles) to amortize staging 4x.
// ---------------------------------------------------------------------------

#define D_IN 128
#define HID  64
#define N_EXP 8
#define CHUNK 64                 // atoms per chunk (4 waves x 16)
#define CHUNKS_PER_BLOCK 4
#define REGION_PAD 256           // CHUNK * CHUNKS_PER_BLOCK

#define W1T_STRIDE 136   // 128 + 8 halves pad (272B row = 16B aligned, bank-spread)
#define W2T_STRIDE 72    // 64 + 8
#define H_STRIDE   72    // 64 + 8

typedef _Float16 v8h  __attribute__((ext_vector_type(8)));
typedef _Float16 v16h __attribute__((ext_vector_type(16)));
typedef float    v8f  __attribute__((ext_vector_type(8)));
typedef float    v4f  __attribute__((ext_vector_type(4)));

__device__ __forceinline__ v16h cat16(v8h a, v8h b) {
  return __builtin_shufflevector(a, b, 0,1,2,3,4,5,6,7,8,9,10,11,12,13,14,15);
}

__device__ __forceinline__ void cvt8(v4f a, v4f b, v8h& hi, v8h& lo) {
#pragma unroll
  for (int i = 0; i < 4; ++i) {
    float x = a[i]; _Float16 h = (_Float16)x;
    hi[i] = h; lo[i] = (_Float16)(x - (float)h);
  }
#pragma unroll
  for (int i = 0; i < 4; ++i) {
    float x = b[i]; _Float16 h = (_Float16)x;
    hi[i + 4] = h; lo[i + 4] = (_Float16)(x - (float)h);
  }
}

__device__ __forceinline__ int z2e(int z) {
  // Z in {1,6,7,8,9,15,16,17} -> expert {0..7}
  if (z == 1) return 0;
  if (z < 10) return z - 5;   // 6,7,8,9 -> 1,2,3,4
  return z - 10;              // 15,16,17 -> 5,6,7
}

// ws layout (ints): [0..7] counts | [8..16] padded offsets | [20..27] cursors
// [32 ..) perm (N + up to 8*REGION_PAD padded slots, padding marked -1)

__global__ void en_init_kernel(int* ws) {
  if (threadIdx.x < 32) ws[threadIdx.x] = 0;
}

__global__ void en_count_kernel(const int* __restrict__ Z, int N, int* __restrict__ ws) {
  __shared__ int lc[N_EXP];
  int tid = threadIdx.x;
  if (tid < N_EXP) lc[tid] = 0;
  __syncthreads();
  int i = blockIdx.x * blockDim.x + tid;
  if (i < N) atomicAdd(&lc[z2e(Z[i])], 1);
  __syncthreads();
  if (tid < N_EXP && lc[tid] > 0) atomicAdd(&ws[tid], lc[tid]);
}

__global__ void en_offsets_kernel(int* ws) {
  __shared__ int soff[N_EXP + 1];
  __shared__ int scnt[N_EXP];
  int* counts = ws;
  int* off    = ws + 8;
  int* cur    = ws + 20;
  int* perm   = ws + 32;
  if (threadIdx.x == 0) {
    int o = 0;
    for (int e = 0; e < N_EXP; ++e) {
      int c = counts[e];
      scnt[e] = c;
      soff[e] = o; off[e] = o; cur[e] = o;
      o += (c + REGION_PAD - 1) & ~(REGION_PAD - 1);   // pad region to 256 atoms
    }
    soff[N_EXP] = o; off[N_EXP] = o;
  }
  __syncthreads();
  // mark padding slots invalid
  for (int e = 0; e < N_EXP; ++e) {
    int start = soff[e] + scnt[e], end = soff[e + 1];
    for (int i = start + threadIdx.x; i < end; i += blockDim.x) perm[i] = -1;
  }
}

__global__ void en_scatter_kernel(const int* __restrict__ Z, int N, int* __restrict__ ws) {
  __shared__ int lc[N_EXP];
  __shared__ int lbase[N_EXP];
  int* cur  = ws + 20;
  int* perm = ws + 32;
  int tid = threadIdx.x;
  if (tid < N_EXP) lc[tid] = 0;
  __syncthreads();
  int i = blockIdx.x * blockDim.x + tid;
  int e = 0, rank = 0;
  if (i < N) {
    e = z2e(Z[i]);
    rank = atomicAdd(&lc[e], 1);
  }
  __syncthreads();
  if (tid < N_EXP && lc[tid] > 0) lbase[tid] = atomicAdd(&cur[tid], lc[tid]);
  __syncthreads();
  if (i < N) perm[lbase[e] + rank] = i;
}

// ---------------------------------------------------------------------------
// Main kernel: 128 threads = 4 waves; each wave owns one 16-atom tile per
// chunk; 4 chunks of 64 atoms per block, one expert per block.
// ---------------------------------------------------------------------------
__global__ void __launch_bounds__(128)
en_moe_tile_kernel(const float* __restrict__ X,
                   const int*  __restrict__ perm,
                   const int*  __restrict__ padoff,
                   const float* __restrict__ W1, const float* __restrict__ b1,
                   const float* __restrict__ W2, const float* __restrict__ b2,
                   const float* __restrict__ W3, const float* __restrict__ b3,
                   float* __restrict__ out)
{
  __shared__ _Float16 sW1h[HID * W1T_STRIDE];   // W1^T hi  [n][k]  17408 B
  __shared__ _Float16 sW1l[HID * W1T_STRIDE];   // W1^T lo          17408 B
  __shared__ _Float16 sW2h[HID * W2T_STRIDE];   // W2^T hi           9216 B
  __shared__ _Float16 sHh[4 * 16 * H_STRIDE];   // h1 hi per wave    9216 B
  __shared__ _Float16 sHl[4 * 16 * H_STRIDE];   // h1 lo             9216 B
  __shared__ float sB1[HID], sB2[HID], sW3[HID];
  __shared__ float sB3;                         // total ~63.2 KB

  const int rb_block = blockIdx.x * REGION_PAD;
  if (rb_block >= padoff[8]) return;
  int e = 0;
#pragma unroll
  for (int k = 1; k < N_EXP; ++k) if (padoff[k] <= rb_block) e = k;

  const int tid = threadIdx.x;

  // ---- stage expert weights into LDS (transposed, split hi/lo f16) ----
  const float* W1e = W1 + e * (D_IN * HID);     // [din][hid] row-major
#pragma unroll
  for (int j = 0; j < 16; ++j) {
    int idx = j * 512 + tid * 4;
    v4f v = *(const v4f*)(W1e + idx);
    int din = idx >> 6, hid = idx & 63;
#pragma unroll
    for (int i = 0; i < 4; ++i) {
      float x = v[i]; _Float16 h = (_Float16)x;
      sW1h[(hid + i) * W1T_STRIDE + din] = h;
      sW1l[(hid + i) * W1T_STRIDE + din] = (_Float16)(x - (float)h);
    }
  }
  const float* W2e = W2 + e * (HID * HID);      // [k][hid]
#pragma unroll
  for (int j = 0; j < 8; ++j) {
    int idx = j * 512 + tid * 4;
    v4f v = *(const v4f*)(W2e + idx);
    int kk = idx >> 6, hid = idx & 63;
#pragma unroll
    for (int i = 0; i < 4; ++i)
      sW2h[(hid + i) * W2T_STRIDE + kk] = (_Float16)v[i];
  }
  if (tid < HID) {
    sB1[tid] = b1[e * HID + tid];
    sB2[tid] = b2[e * HID + tid];
    sW3[tid] = W3[e * HID + tid];
  }
  if (tid == 0) sB3 = b3[e];
  __syncthreads();

  const int wave   = tid >> 5;
  const int lane   = tid & 31;
  const int lane15 = lane & 15;
  const int laneHi = lane >> 4;
  _Float16* hh = &sHh[wave * 16 * H_STRIDE];
  _Float16* hl = &sHl[wave * 16 * H_STRIDE];

#pragma unroll 1
  for (int c = 0; c < CHUNKS_PER_BLOCK; ++c) {
    const int tile_base = rb_block + c * CHUNK + wave * 16;
    const int atom = perm[tile_base + lane15];  // row M = lane15 of this tile

    // prefetch next chunk's X rows while we compute this one
    if (c + 1 < CHUNKS_PER_BLOCK) {
      int atom_n = perm[tile_base + CHUNK + lane15];
      if (atom_n >= 0) {
        const char* p = (const char*)(X + (size_t)atom_n * D_IN) + laneHi * 256;
        __builtin_prefetch(p, 0, 3);
        __builtin_prefetch(p + 128, 0, 3);
      }
    }

    // ---- layer 1: [16x128] x [128x64], split f16 3-term ----
    v8f acc[4] = {};
#pragma unroll
    for (int t = 0; t < 4; ++t) {               // K blocks of 32
      int colbase = t * 32 + laneHi * 8;
      v8h ah1, al1, ah2, al2;
      if (atom >= 0) {
        const float* xr = X + (size_t)atom * D_IN + colbase;
        v4f c0 = *(const v4f*)(xr);
        v4f c1 = *(const v4f*)(xr + 4);
        v4f c2 = *(const v4f*)(xr + 16);
        v4f c3 = *(const v4f*)(xr + 20);
        cvt8(c0, c1, ah1, al1);
        cvt8(c2, c3, ah2, al2);
      } else {
#pragma unroll
        for (int i = 0; i < 8; ++i) { ah1[i] = (_Float16)0.f; al1[i] = (_Float16)0.f;
                                      ah2[i] = (_Float16)0.f; al2[i] = (_Float16)0.f; }
      }
      v16h aH = cat16(ah1, ah2);                // A layout: K {0-7,16-23} / {8-15,24-31}
      v16h aL = cat16(al1, al2);
#pragma unroll
      for (int nt = 0; nt < 4; ++nt) {          // N tiles of 16
        int n = nt * 16 + lane15;
        const _Float16* bh = &sW1h[n * W1T_STRIDE + t * 32 + laneHi * 8];
        const _Float16* bl = &sW1l[n * W1T_STRIDE + t * 32 + laneHi * 8];
        v16h bH = cat16(*(const v8h*)bh, *(const v8h*)(bh + 16));  // B: K {0-7,16-23} / {8-15,24-31}
        v16h bL = cat16(*(const v8h*)bl, *(const v8h*)(bl + 16));
        acc[nt] = __builtin_amdgcn_wmma_f32_16x16x32_f16(false, aH, false, bH, (short)0, acc[nt], false, false);
        acc[nt] = __builtin_amdgcn_wmma_f32_16x16x32_f16(false, aL, false, bH, (short)0, acc[nt], false, false);
        acc[nt] = __builtin_amdgcn_wmma_f32_16x16x32_f16(false, aH, false, bL, (short)0, acc[nt], false, false);
      }
    }

    // ---- bias + silu, re-split to hi/lo f16 in LDS (per-wave region) ----
#pragma unroll
    for (int nt = 0; nt < 4; ++nt) {
      int n = nt * 16 + lane15;
      float bb = sB1[n];
#pragma unroll
      for (int r = 0; r < 8; ++r) {             // D: row = r + 8*laneHi, col = n
        float x = acc[nt][r] + bb;
        float s = x / (1.f + __expf(-x));
        _Float16 h = (_Float16)s;
        int row = r + laneHi * 8;
        hh[row * H_STRIDE + n] = h;
        hl[row * H_STRIDE + n] = (_Float16)(s - (float)h);
      }
    }
    __builtin_amdgcn_wave_barrier();            // wave-private LDS; keep order

    // ---- layer 2: [16x64] x [64x64], split-h 2-term ----
    v8f acc2[4] = {};
#pragma unroll
    for (int t = 0; t < 2; ++t) {
      int colbase = t * 32 + laneHi * 8;
      const _Float16* ar = &hh[lane15 * H_STRIDE + colbase];
      const _Float16* al = &hl[lane15 * H_STRIDE + colbase];
      v16h aH = cat16(*(const v8h*)ar, *(const v8h*)(ar + 16));
      v16h aL = cat16(*(const v8h*)al, *(const v8h*)(al + 16));
#pragma unroll
      for (int nt = 0; nt < 4; ++nt) {
        int n = nt * 16 + lane15;
        const _Float16* bp = &sW2h[n * W2T_STRIDE + t * 32 + laneHi * 8];
        v16h bH = cat16(*(const v8h*)bp, *(const v8h*)(bp + 16));
        acc2[nt] = __builtin_amdgcn_wmma_f32_16x16x32_f16(false, aH, false, bH, (short)0, acc2[nt], false, false);
        acc2[nt] = __builtin_amdgcn_wmma_f32_16x16x32_f16(false, aL, false, bH, (short)0, acc2[nt], false, false);
      }
    }

    // ---- layer 3: silu, dot with W3, 16-lane butterfly reduce ----
    float partial[8];
#pragma unroll
    for (int r = 0; r < 8; ++r) partial[r] = 0.f;
#pragma unroll
    for (int nt = 0; nt < 4; ++nt) {
      int n = nt * 16 + lane15;
      float bb = sB2[n], w3 = sW3[n];
#pragma unroll
      for (int r = 0; r < 8; ++r) {
        float x = acc2[nt][r] + bb;
        float s = x / (1.f + __expf(-x));
        partial[r] += s * w3;
      }
    }
#pragma unroll
    for (int r = 0; r < 8; ++r) {
      partial[r] += __shfl_xor(partial[r], 1, 32);
      partial[r] += __shfl_xor(partial[r], 2, 32);
      partial[r] += __shfl_xor(partial[r], 4, 32);
      partial[r] += __shfl_xor(partial[r], 8, 32);
    }
    if (lane15 == 0) {                          // lane 0 -> rows 0..7, lane 16 -> rows 8..15
      float bb = sB3;
#pragma unroll
      for (int r = 0; r < 8; ++r) {
        int a = perm[tile_base + r + laneHi * 8];
        if (a >= 0) out[a] = partial[r] + bb;
      }
    }
    __builtin_amdgcn_wave_barrier();            // keep h-tile reuse ordered across chunks
  }
}

extern "C" void kernel_launch(void* const* d_in, const int* in_sizes, int n_in,
                              void* d_out, int out_size, void* d_ws, size_t ws_size,
                              hipStream_t stream) {
  const float* X  = (const float*)d_in[0];
  const int*   Z  = (const int*)  d_in[1];
  const float* W1 = (const float*)d_in[2];
  const float* b1 = (const float*)d_in[3];
  const float* W2 = (const float*)d_in[4];
  const float* b2 = (const float*)d_in[5];
  const float* W3 = (const float*)d_in[6];
  const float* b3 = (const float*)d_in[7];
  float* out = (float*)d_out;

  const int N = in_sizes[1];          // one Z per atom
  int* ws = (int*)d_ws;

  en_init_kernel<<<1, 32, 0, stream>>>(ws);
  int cb = (N + 255) / 256;
  en_count_kernel<<<cb, 256, 0, stream>>>(Z, N, ws);
  en_offsets_kernel<<<1, 64, 0, stream>>>(ws);
  en_scatter_kernel<<<cb, 256, 0, stream>>>(Z, N, ws);

  int mb = (N + REGION_PAD - 1) / REGION_PAD + N_EXP;  // covers padded regions
  en_moe_tile_kernel<<<mb, 128, 0, stream>>>(X, ws + 32, ws + 8,
                                             W1, b1, W2, b2, W3, b3, out);
}
